// GraphNeuralDecoder_88613765251355
// MI455X (gfx1250) — matrix operand order repaired and checked by reading, not compiled, not run on hardware
//
#include <hip/hip_runtime.h>
#include <cmath>

// ---------------------------------------------------------------------------
// GraphNeuralDecoder, fully fused for MI455X (gfx1250, wave32, WMMA bf16)
//   B=4096, T=32, NC=32, D=256, V=256
// One workgroup (256 threads = 8 waves) per batch element. All activations
// live in LDS; GEMMs use v_wmma_f32_16x16x32_bf16 with f32 accumulation.
// Round-2 fix: rolled K-loop (no B-frag spills), A-frag reuse across N-tiles,
// vectorized bf16 conversion glue.
// ---------------------------------------------------------------------------

typedef __bf16 v16bf  __attribute__((ext_vector_type(16)));
typedef __bf16 bf16x8 __attribute__((ext_vector_type(8)));
typedef __bf16 bf16x4 __attribute__((ext_vector_type(4)));
typedef float  v8f    __attribute__((ext_vector_type(8)));
typedef float  f32x4  __attribute__((ext_vector_type(4)));

union AFrag { v16bf v; bf16x8 h[2]; };

#define D_DIM   256
#define T_ROWS  32
#define CAT_K   512
#define LN_EPS  1e-5f

// LDS carve offsets (bytes)
#define OFF_TOKF   0         // 32*256*4 = 32768  fp32 token master
#define OFF_CELLF  32768     // 32768             fp32 cell master
#define OFF_MSGT   65536     // 32768             fp32 token message accum
#define OFF_MSGC   98304     // 32768             fp32 cell message accum
#define OFF_CAT    131072    // 32*512*2 = 32768  bf16 concat GEMM input
#define OFF_HB     163840    // 32*256*2 = 16384  bf16 hidden / LN output
#define OFF_MEANT  180224    // 1024              fp32 token mean
#define OFF_MEANC  181248    // 1024              fp32 cell mean
#define SHMEM_BYTES 182272

__device__ __forceinline__ float gelu_exact(float x) {
    return 0.5f * x * (1.0f + erff(x * 0.70710678118654752f));
}

// Epilogue modes
#define EP_GELU_BF16   0   // gelu -> bf16 LDS (hidden)
#define EP_MSG_INIT    1   // -> f32 LDS (msg =)
#define EP_MSG_ACC     2   // -> f32 LDS (msg +=)
#define EP_GELU_F32    3   // gelu -> f32 LDS (master update)
#define EP_STORE_GLOB  4   // -> f32 global (logits)

// GEMM: Y[32][256] = A[32][KDIM] (bf16, LDS row-major) @ Bpacked[KDIM][256]
// Wave w owns N-tiles {2w, 2w+1}. Per K-step: one A-frag pair (both M-tiles),
// two B-frags, 4 WMMAs. K-loop kept rolled (unroll 2) to avoid VGPR spills.
template<int KDIM, int MODE>
__device__ __forceinline__ void wmma_gemm(
    const __bf16* __restrict__ A,     // LDS, row-major [32][KDIM]
    const __bf16* __restrict__ Bp,    // global, tile-packed bf16
    const float*  __restrict__ bias,  // global fp32, may be nullptr
    __bf16* __restrict__ outB,        // LDS bf16 [32][256]
    float*  __restrict__ outF,        // LDS f32  [32][256]
    float*  __restrict__ outG,        // global f32 (pre-offset)
    int lane, int wave)
{
    constexpr int KT = KDIM / 32;
    const int lg  = lane >> 4;   // lane group (0/1)
    const int ln  = lane & 15;
    const int nt0 = wave * 2;

    v8f acc00 = {0.f,0.f,0.f,0.f,0.f,0.f,0.f,0.f};  // nt0,   mt0
    v8f acc01 = {0.f,0.f,0.f,0.f,0.f,0.f,0.f,0.f};  // nt0,   mt1
    v8f acc10 = {0.f,0.f,0.f,0.f,0.f,0.f,0.f,0.f};  // nt0+1, mt0
    v8f acc11 = {0.f,0.f,0.f,0.f,0.f,0.f,0.f,0.f};  // nt0+1, mt1

    const int arow0 = ln;        // M-tile 0 row
    const int arow1 = 16 + ln;   // M-tile 1 row

#pragma unroll 2
    for (int kt = 0; kt < KT; ++kt) {
        // A fragments: lanes 0-15 hold K = 0..15; lanes 16-31 hold K = 16..31
        const int ab = kt * 32 + lg * 16;
        AFrag a0, a1;
        a0.h[0] = *(const bf16x8*)&A[arow0 * KDIM + ab];
        a0.h[1] = *(const bf16x8*)&A[arow0 * KDIM + ab + 8];
        a1.h[0] = *(const bf16x8*)&A[arow1 * KDIM + ab];
        a1.h[1] = *(const bf16x8*)&A[arow1 * KDIM + ab + 8];
        // B fragments for both owned N-tiles (adjacent packed tiles)
        const __bf16* bbase = Bp + (((size_t)kt * 16 + nt0) * 32 + lane) * 16;
        v16bf b0 = *(const v16bf*)bbase;
        v16bf b1 = *(const v16bf*)(bbase + 512);
        if (kt + 1 < KT) {
            __builtin_prefetch(Bp + (((size_t)(kt + 1) * 16 + nt0) * 32 + lane) * 16, 0, 1);
        }
        acc00 = __builtin_amdgcn_wmma_f32_16x16x32_bf16(
            false, a0.v, false, b0, (short)0, acc00, false, false);
        acc01 = __builtin_amdgcn_wmma_f32_16x16x32_bf16(
            false, a1.v, false, b0, (short)0, acc01, false, false);
        acc10 = __builtin_amdgcn_wmma_f32_16x16x32_bf16(
            false, a0.v, false, b1, (short)0, acc10, false, false);
        acc11 = __builtin_amdgcn_wmma_f32_16x16x32_bf16(
            false, a1.v, false, b1, (short)0, acc11, false, false);
    }

    // Epilogue: C/D layout — VGPR r: lanes 0-15 M=r, lanes 16-31 M=r+8
#pragma unroll
    for (int j = 0; j < 2; ++j) {
        const int col = (nt0 + j) * 16 + ln;
        const float bv = bias ? bias[col] : 0.f;
        const v8f aA = j ? acc10 : acc00;   // mt0
        const v8f aB = j ? acc11 : acc01;   // mt1
#pragma unroll
        for (int r = 0; r < 8; ++r) {
            const int row0 = r + lg * 8;
            const int row1 = 16 + row0;
            const float v0 = aA[r] + bv;
            const float v1 = aB[r] + bv;
            if (MODE == EP_GELU_BF16) {
                outB[row0 * 256 + col] = (__bf16)gelu_exact(v0);
                outB[row1 * 256 + col] = (__bf16)gelu_exact(v1);
            } else if (MODE == EP_MSG_INIT) {
                outF[row0 * 256 + col] = v0;
                outF[row1 * 256 + col] = v1;
            } else if (MODE == EP_MSG_ACC) {
                outF[row0 * 256 + col] += v0;
                outF[row1 * 256 + col] += v1;
            } else if (MODE == EP_GELU_F32) {
                outF[row0 * 256 + col] = gelu_exact(v0);
                outF[row1 * 256 + col] = gelu_exact(v1);
            } else {
                outG[row0 * 256 + col] = v0;
                outG[row1 * 256 + col] = v1;
            }
        }
    }
}

// Build bf16 concat input [32][512] = [first(fp32 LDS [32][256]) | second]
// second is either a 256-vector broadcast (rowwise=false) or [32][256] rowwise.
// Vectorized: 4 elements per op.
__device__ __forceinline__ void build_cat(__bf16* __restrict__ cat,
                                          const float* __restrict__ first,
                                          const float* __restrict__ second,
                                          bool rowwise, int tid)
{
    for (int q = tid; q < (32 * 512) / 4; q += 256) {   // 16 iterations
        const int row = q >> 7;
        const int c   = (q & 127) * 4;
        f32x4 v;
        if (c < 256)      v = *(const f32x4*)&first[row * 256 + c];
        else if (rowwise) v = *(const f32x4*)&second[row * 256 + (c - 256)];
        else              v = *(const f32x4*)&second[c - 256];
        bf16x4 o;
        o.x = (__bf16)v.x; o.y = (__bf16)v.y; o.z = (__bf16)v.z; o.w = (__bf16)v.w;
        *(bf16x4*)&cat[row * 512 + c] = o;
    }
}

// ---------------------------------------------------------------------------
// Prep kernel 1: repack fp32 weight [K][N] into bf16 WMMA-B tile order.
// Packed element idx = ((kt*NT + nt)*32 + lane)*16 + e, with
//   k = kt*32 + (lane>>4)*16 + e ; n = nt*16 + (lane&15)
// so each lane's 16 bf16 are contiguous-K column slices (one v16bf load).
// ---------------------------------------------------------------------------
__global__ void pack_weight_kernel(const float* __restrict__ src,
                                   __bf16* __restrict__ dst, int K, int N)
{
    const int idx = blockIdx.x * 256 + threadIdx.x;
    if (idx >= K * N) return;
    const int e    = idx & 15;
    const int lane = (idx >> 4) & 31;
    const int tile = idx >> 9;
    const int NT   = N >> 4;
    const int kt   = tile / NT;
    const int nt   = tile % NT;
    const int k = kt * 32 + ((lane >> 4) << 4) + e;
    const int n = nt * 16 + (lane & 15);
    dst[idx] = (__bf16)src[k * N + n];
}

// Prep kernel 2: cell0 = c_states @ cellp_w + cellp_b   (32x256, shared by all b)
__global__ void cellproj_kernel(const float* __restrict__ c_states,
                                const float* __restrict__ w,
                                const float* __restrict__ bvec,
                                float* __restrict__ out)
{
    const int j = threadIdx.x;   // 0..255
    const int i = blockIdx.x;    // 0..31
    float s = bvec[j];
    for (int k = 0; k < 256; ++k)
        s += c_states[i * 256 + k] * w[k * 256 + j];
    out[i * 256 + j] = s;
}

// ---------------------------------------------------------------------------
// Main fused kernel: one workgroup per batch element.
// ---------------------------------------------------------------------------
__global__ __launch_bounds__(256)
void gnd_main_kernel(const int* __restrict__ tokens,
                     const float* __restrict__ embed_w,
                     const float* __restrict__ pos_w,
                     const float* __restrict__ cell0,
                     const __bf16* __restrict__ p_tt1, const float* __restrict__ tt_b1,
                     const __bf16* __restrict__ p_tt2, const float* __restrict__ tt_b2,
                     const __bf16* __restrict__ p_tc1, const float* __restrict__ tc_b1,
                     const __bf16* __restrict__ p_tc2, const float* __restrict__ tc_b2,
                     const __bf16* __restrict__ p_cc1, const float* __restrict__ cc_b1,
                     const __bf16* __restrict__ p_cc2, const float* __restrict__ cc_b2,
                     const __bf16* __restrict__ p_ut,  const float* __restrict__ ut_b,
                     const __bf16* __restrict__ p_uc,  const float* __restrict__ uc_b,
                     const float* __restrict__ ln_g, const float* __restrict__ ln_b,
                     const __bf16* __restrict__ p_head,
                     float* __restrict__ out)
{
    extern __shared__ __align__(16) char smem[];
    float*  tokF  = (float*)(smem + OFF_TOKF);
    float*  cellF = (float*)(smem + OFF_CELLF);
    float*  msgT  = (float*)(smem + OFF_MSGT);
    float*  msgC  = (float*)(smem + OFF_MSGC);
    __bf16* cat   = (__bf16*)(smem + OFF_CAT);
    __bf16* hB    = (__bf16*)(smem + OFF_HB);
    float*  meanT = (float*)(smem + OFF_MEANT);
    float*  meanC = (float*)(smem + OFF_MEANC);

    const int tid  = threadIdx.x;
    const int wave = tid >> 5;
    const int lane = tid & 31;
    const int b    = blockIdx.x;

    // ---- init: tok = embed[tokens] + pos ; cell = broadcast(cell0) ----
    const int* toks = tokens + b * T_ROWS;
    for (int q = tid; q < (32 * 256) / 4; q += 256) {   // 8 iterations, 4-wide
        const int row = q >> 6;
        const int col = (q & 63) * 4;
        const int tk = toks[row];
        const f32x4 e  = *(const f32x4*)&embed_w[tk * 256 + col];
        const f32x4 p  = *(const f32x4*)&pos_w[row * 256 + col];
        const f32x4 c0 = *(const f32x4*)&cell0[row * 256 + col];
        *(f32x4*)&tokF[row * 256 + col]  = e + p;
        *(f32x4*)&cellF[row * 256 + col] = c0;
    }
    __syncthreads();

    // ---- 2 rounds of message passing ----
    for (int round = 0; round < 2; ++round) {
        // means over rows (tid == column index)
        {
            float st = 0.f, sc = 0.f;
            for (int r = 0; r < 32; ++r) {
                st += tokF[r * 256 + tid];
                sc += cellF[r * 256 + tid];
            }
            meanT[tid] = st * (1.f / 32.f);
            meanC[tid] = sc * (1.f / 32.f);
        }
        __syncthreads();

        // msg_tt = MLP_tt([tok | mean(tok)])      -> msgT (init)
        build_cat(cat, tokF, meanT, false, tid);                 __syncthreads();
        wmma_gemm<CAT_K, EP_GELU_BF16>(cat, p_tt1, tt_b1, hB, nullptr, nullptr, lane, wave); __syncthreads();
        wmma_gemm<D_DIM, EP_MSG_INIT>(hB, p_tt2, tt_b2, nullptr, msgT, nullptr, lane, wave); __syncthreads();

        // msg_tc = MLP_tc([tok | mean(cell)])     -> msgT (accum)
        build_cat(cat, tokF, meanC, false, tid);                 __syncthreads();
        wmma_gemm<CAT_K, EP_GELU_BF16>(cat, p_tc1, tc_b1, hB, nullptr, nullptr, lane, wave); __syncthreads();
        wmma_gemm<D_DIM, EP_MSG_ACC>(hB, p_tc2, tc_b2, nullptr, msgT, nullptr, lane, wave); __syncthreads();

        // msg_cc = MLP_cc([cell | mean(cell)])    -> msgC (init)
        build_cat(cat, cellF, meanC, false, tid);                __syncthreads();
        wmma_gemm<CAT_K, EP_GELU_BF16>(cat, p_cc1, cc_b1, hB, nullptr, nullptr, lane, wave); __syncthreads();
        wmma_gemm<D_DIM, EP_MSG_INIT>(hB, p_cc2, cc_b2, nullptr, msgC, nullptr, lane, wave); __syncthreads();

        // msg_ct = MLP_tc([cell | mean(tok)])     -> msgC (accum)  (tc weight reuse)
        build_cat(cat, cellF, meanT, false, tid);                __syncthreads();
        wmma_gemm<CAT_K, EP_GELU_BF16>(cat, p_tc1, tc_b1, hB, nullptr, nullptr, lane, wave); __syncthreads();
        wmma_gemm<D_DIM, EP_MSG_ACC>(hB, p_tc2, tc_b2, nullptr, msgC, nullptr, lane, wave); __syncthreads();

        // tok  = gelu([tok  | msgT] @ ut_w + ut_b)
        build_cat(cat, tokF, msgT, true, tid);                   __syncthreads();
        wmma_gemm<CAT_K, EP_GELU_F32>(cat, p_ut, ut_b, nullptr, tokF, nullptr, lane, wave); __syncthreads();

        // cell = gelu([cell | msgC] @ uc_w + uc_b)
        build_cat(cat, cellF, msgC, true, tid);                  __syncthreads();
        wmma_gemm<CAT_K, EP_GELU_F32>(cat, p_uc, uc_b, nullptr, cellF, nullptr, lane, wave); __syncthreads();
    }

    // ---- LayerNorm rows of tok -> hB (bf16) ----
    // wave handles 4 rows; lane-parallel reduce over 256 cols (wave32 shuffles)
    for (int rr = 0; rr < 4; ++rr) {
        const int row = wave * 4 + rr;
        float s1 = 0.f, s2 = 0.f;
#pragma unroll
        for (int i = 0; i < 8; ++i) {
            const float x = tokF[row * 256 + lane + i * 32];
            s1 += x; s2 += x * x;
        }
#pragma unroll
        for (int off = 16; off > 0; off >>= 1) {
            s1 += __shfl_xor(s1, off, 32);
            s2 += __shfl_xor(s2, off, 32);
        }
        const float mu  = s1 * (1.f / 256.f);
        const float var = s2 * (1.f / 256.f) - mu * mu;
        const float inv = rsqrtf(var + LN_EPS);
#pragma unroll
        for (int i = 0; i < 8; ++i) {
            const int c = lane + i * 32;
            const float xn = (tokF[row * 256 + c] - mu) * inv;
            hB[row * 256 + c] = (__bf16)(xn * ln_g[c] + ln_b[c]);
        }
    }
    __syncthreads();

    // ---- head: logits[32][256] = xn @ head_w -> global ----
    float* outG = out + (size_t)b * T_ROWS * 256;
    wmma_gemm<D_DIM, EP_STORE_GLOB>(hB, p_head, nullptr, nullptr, nullptr, outG, lane, wave);
}

// ---------------------------------------------------------------------------
// Host launcher
// ---------------------------------------------------------------------------
extern "C" void kernel_launch(void* const* d_in, const int* in_sizes, int n_in,
                              void* d_out, int out_size, void* d_ws, size_t ws_size,
                              hipStream_t stream)
{
    const int*   tokens   = (const int*)  d_in[0];
    const float* c_states = (const float*)d_in[1];
    const float* embed_w  = (const float*)d_in[2];
    const float* pos_w    = (const float*)d_in[3];
    const float* cellp_w  = (const float*)d_in[4];
    const float* cellp_b  = (const float*)d_in[5];
    const float* tt_w1 = (const float*)d_in[6];  const float* tt_b1 = (const float*)d_in[7];
    const float* tt_w2 = (const float*)d_in[8];  const float* tt_b2 = (const float*)d_in[9];
    const float* tc_w1 = (const float*)d_in[10]; const float* tc_b1 = (const float*)d_in[11];
    const float* tc_w2 = (const float*)d_in[12]; const float* tc_b2 = (const float*)d_in[13];
    const float* cc_w1 = (const float*)d_in[14]; const float* cc_b1 = (const float*)d_in[15];
    const float* cc_w2 = (const float*)d_in[16]; const float* cc_b2 = (const float*)d_in[17];
    const float* ut_w  = (const float*)d_in[18]; const float* ut_b  = (const float*)d_in[19];
    const float* uc_w  = (const float*)d_in[20]; const float* uc_b  = (const float*)d_in[21];
    const float* ln_g  = (const float*)d_in[22];
    const float* ln_b  = (const float*)d_in[23];
    const float* head_w = (const float*)d_in[24];
    float* out = (float*)d_out;

    // workspace: cell0 fp32 [32*256], then bf16 packed weights
    char* ws = (char*)d_ws;
    float*  cell0 = (float*)ws;
    __bf16* pw    = (__bf16*)(ws + 32 * 256 * 4);
    __bf16* p_tt1 = pw;            // 512*256
    __bf16* p_tt2 = pw + 131072;   // 256*256
    __bf16* p_tc1 = pw + 196608;   // 512*256
    __bf16* p_tc2 = pw + 327680;   // 256*256
    __bf16* p_cc1 = pw + 393216;   // 512*256
    __bf16* p_cc2 = pw + 524288;   // 256*256
    __bf16* p_ut  = pw + 589824;   // 512*256
    __bf16* p_uc  = pw + 720896;   // 512*256
    __bf16* p_head= pw + 851968;   // 256*256

    auto pack = [&](const float* src, __bf16* dst, int K, int N) {
        const int total = K * N;
        pack_weight_kernel<<<(total + 255) / 256, 256, 0, stream>>>(src, dst, K, N);
    };
    pack(tt_w1, p_tt1, 512, 256);
    pack(tt_w2, p_tt2, 256, 256);
    pack(tc_w1, p_tc1, 512, 256);
    pack(tc_w2, p_tc2, 256, 256);
    pack(cc_w1, p_cc1, 512, 256);
    pack(cc_w2, p_cc2, 256, 256);
    pack(ut_w,  p_ut,  512, 256);
    pack(uc_w,  p_uc,  512, 256);
    pack(head_w, p_head, 256, 256);

    cellproj_kernel<<<32, 256, 0, stream>>>(c_states, cellp_w, cellp_b, cell0);

    gnd_main_kernel<<<4096, 256, SHMEM_BYTES, stream>>>(
        tokens, embed_w, pos_w, cell0,
        p_tt1, tt_b1, p_tt2, tt_b2,
        p_tc1, tc_b1, p_tc2, tc_b2,
        p_cc1, cc_b1, p_cc2, cc_b2,
        p_ut, ut_b, p_uc, uc_b,
        ln_g, ln_b, p_head, out);
}
